// EnhancedGCNII_28759101014307
// MI455X (gfx1250) — matrix operand order repaired and checked by reading, not compiled, not run on hardware
//
#include <hip/hip_runtime.h>
#include <hip/hip_bf16.h>
#include <math.h>

typedef __attribute__((ext_vector_type(16))) _Float16 v16h;
typedef __attribute__((ext_vector_type(8)))  float    v8f;
typedef __attribute__((ext_vector_type(4)))  int      v4i;

#define NROW   8192
#define NFEAT  500
#define NFEATP 512
#define NHID   128
#define NCLASS 40
#define NLAYERS 4
#define ALPHA_C 0.1f
#define GAMMA_C 0.1f

// gfx1250 async global->LDS path (ASYNCcnt-tracked DMA, no VGPR staging).
// Guarded: falls back to synchronous copies if the builtins are absent.
#if defined(__gfx1250__) && __has_builtin(__builtin_amdgcn_global_load_async_to_lds_b128) && \
    __has_builtin(__builtin_amdgcn_s_wait_asynccnt)
#define HAVE_ASYNC_LDS 1
#else
#define HAVE_ASYNC_LDS 0
#endif

__device__ __forceinline__ void cp_b128_to_lds(const _Float16* g, _Float16* l) {
#if HAVE_ASYNC_LDS
  // signature (from compiler diagnostic): (v4i addrspace(1)*, v4i addrspace(3)*, imm offset, imm cpol)
  __builtin_amdgcn_global_load_async_to_lds_b128(
      (__attribute__((address_space(1))) v4i*)g,
      (__attribute__((address_space(3))) v4i*)l, 0, 0);
#else
  *(uint4*)l = *(const uint4*)g;
#endif
}
__device__ __forceinline__ void wait_async_le4() {
#if HAVE_ASYNC_LDS
  __builtin_amdgcn_s_wait_asynccnt(4);
#endif
}
__device__ __forceinline__ void wait_async_le0() {
#if HAVE_ASYNC_LDS
  __builtin_amdgcn_s_wait_asynccnt(0);
#endif
}

// ---------------- degree / D^{-1/2} ----------------
__global__ void k_degree(const float* __restrict__ adj, float* __restrict__ dinv) {
  int row = blockIdx.x, tid = threadIdx.x;
  const float4* p = (const float4*)(adj + (size_t)row * NROW);
  float s = 0.f;
  for (int i = tid; i < NROW / 4; i += 256) { float4 v = p[i]; s += v.x + v.y + v.z + v.w; }
  __shared__ float red[256];
  red[tid] = s; __syncthreads();
  for (int off = 128; off > 0; off >>= 1) {
    if (tid < off) red[tid] += red[tid + off];
    __syncthreads();
  }
  if (tid == 0) dinv[row] = rsqrtf(red[0] + 1.0f);   // +1 self loop, deg>=1
}

// ---------------- a_hat = D^-1/2 (A+I) D^-1/2 in f16 (128 MB, L2-resident) ----
__global__ void k_build_ahat(const float* __restrict__ adj, const float* __restrict__ dinv,
                             _Float16* __restrict__ ahat) {
  int row  = blockIdx.x;
  int col0 = (blockIdx.y * 256 + threadIdx.x) * 8;
  float di = dinv[row];
  const float* src = adj + (size_t)row * NROW + col0;
  float4 a0 = ((const float4*)src)[0];
  float4 a1 = ((const float4*)src)[1];
  float v[8] = {a0.x, a0.y, a0.z, a0.w, a1.x, a1.y, a1.z, a1.w};
  union { _Float16 h[8]; uint4 q; } u;
#pragma unroll
  for (int j = 0; j < 8; j++) {
    int c = col0 + j;
    float a = v[j] + ((c == row) ? 1.0f : 0.0f);
    u.h[j] = (_Float16)(di * a * dinv[c]);
  }
  *(uint4*)(ahat + (size_t)row * NROW + col0) = u.q;
}

// ---------------- f32 -> f16 convert with zero pad ----------------
__global__ void k_cvt(const float* __restrict__ src, _Float16* __restrict__ dst,
                      int srows, int scols, int drows, int dcols) {
  int idx = blockIdx.x * 256 + threadIdx.x;
  if (idx >= drows * dcols) return;
  int r = idx / dcols, c = idx - r * dcols;
  float v = (r < srows && c < scols) ? src[(size_t)r * scols + c] : 0.0f;
  dst[idx] = (_Float16)v;
}

// ---------------- generic WMMA GEMM, fused epilogues ----------------
// Tile: BM=128, BN=64, BK=64. 256 threads = 8 waves; wave w owns rows [16w,16w+16),
// all 64 cols (4 accumulator tiles; A fragment reused 4x; 2 k-substeps per stage).
// A is double-buffered in LDS and filled by async global->LDS DMA one stage ahead
// (zero VGPRs live across the WMMA block); B (needs transpose) is register-staged.
enum { EPI_FCIN = 0, EPI_HW = 1, EPI_BIG = 2, EPI_SMALL = 3, EPI_OUT = 4 };

template <int EPI>
__global__ __launch_bounds__(256)
void k_gemm(const _Float16* __restrict__ A, int lda,
            const _Float16* __restrict__ B, int ldb, int K,
            float* __restrict__ outF, float* __restrict__ outF2,
            _Float16* __restrict__ outH,
            const float* __restrict__ bias0, const float* __restrict__ bias1,
            const float* __restrict__ aux0, const float* __restrict__ aux1,
            float p0) {
  __shared__ __align__(16) _Float16 As[2][128][72];  // ping-pong [m][k], stride 72 (144B)
  __shared__ __align__(16) _Float16 Bs[64][72];      // transposed: [n][k]

  const int tid   = threadIdx.x;
  const int wid   = tid >> 5;
  const int lane  = tid & 31;
  const int lhalf = lane >> 4;   // which 16-lane half
  const int l16   = lane & 15;
  const int blockRow = blockIdx.x * 128;
  const int blockCol = blockIdx.y * 64;

  // staging thread mapping
  const int arow = tid >> 1, ac = (tid & 1) << 5;    // A: 128 rows x 64 halves, 4x b128/thread
  const int bkr  = tid >> 2, bnb = (tid & 3) << 4;   // B: 64 k-rows x 64 cols, 2x b128/thread

  v8f zero = {0.f, 0.f, 0.f, 0.f, 0.f, 0.f, 0.f, 0.f};
  v8f acc[4];
#pragma unroll
  for (int t = 0; t < 4; t++) acc[t] = zero;

  uint4 br[2];
  const int S = K >> 6;
  // prologue: async-fill A buffer 0, fetch B tile 0 into registers
  {
    const _Float16* g = A + (size_t)(blockRow + arow) * lda + ac;
#pragma unroll
    for (int i = 0; i < 4; i++) cp_b128_to_lds(g + 8 * i, &As[0][arow][ac + 8 * i]);
    const uint4* bsrc = (const uint4*)(B + (size_t)bkr * ldb + blockCol + bnb);
    br[0] = bsrc[0]; br[1] = bsrc[1];
  }

  for (int s = 0; s < S; s++) {
    const int p = s & 1;
    // commit staged B (transposed); previous stage's end barrier protects Bs
    {
      union { uint4 q[2]; _Float16 h[16]; } ub;
      ub.q[0] = br[0]; ub.q[1] = br[1];
#pragma unroll
      for (int j = 0; j < 16; j++) Bs[bnb + j][bkr] = ub.h[j];
    }
    // issue next stage's fills (A async -> LDS[1-p], B -> regs), then ensure
    // the CURRENT A buffer's 4 async copies have retired (in-order completion).
    if (s + 1 < S) {
      const int k1 = (s + 1) << 6;
      const _Float16* g = A + (size_t)(blockRow + arow) * lda + k1 + ac;
#pragma unroll
      for (int i = 0; i < 4; i++) cp_b128_to_lds(g + 8 * i, &As[1 - p][arow][ac + 8 * i]);
      const uint4* bsrc = (const uint4*)(B + (size_t)(k1 + bkr) * ldb + blockCol + bnb);
      br[0] = bsrc[0]; br[1] = bsrc[1];
      wait_async_le4();
    } else {
      wait_async_le0();
    }
    __syncthreads();

    // 2 k-substeps of 32 out of As[p] / Bs
    const int am = (wid << 4) | l16;
#pragma unroll
    for (int ks = 0; ks < 64; ks += 32) {
      // A fragment (16x32 f16): lane holds row M=l16 of its wave strip;
      // k = ks + 8*lhalf + j (j<8), ks + 16 + 8*lhalf + (j-8) (j>=8)
      union { v16h v; uint4 q[2]; } af;
      af.q[0] = *(const uint4*)&As[p][am][ks + (lhalf << 3)];
      af.q[1] = *(const uint4*)&As[p][am][ks + 16 + (lhalf << 3)];
#pragma unroll
      for (int t = 0; t < 4; t++) {
        // B fragment (32x16 f16): lane holds col N=l16; k = ks + 8*lhalf + j (j<8), ks + 16 + 8*lhalf + (j-8) (j>=8)
        union { v16h v; uint4 q[2]; } bf;
        int bn = (t << 4) | l16;
        bf.q[0] = *(const uint4*)&Bs[bn][ks + (lhalf << 3)];
        bf.q[1] = *(const uint4*)&Bs[bn][ks + 16 + (lhalf << 3)];
        acc[t] = __builtin_amdgcn_wmma_f32_16x16x32_f16(
            false, af.v, false, bf.v, (short)0, acc[t], false, false);
      }
    }
    __syncthreads();  // all waves done reading As[p]/Bs before next overwrite
  }

  // epilogue: D layout — VGPR r holds (M = r + 8*lhalf, N = l16)
  const int rowBase = blockRow + (wid << 4) + (lhalf << 3);
#pragma unroll
  for (int t = 0; t < 4; t++) {
    int col = blockCol + (t << 4) + l16;
#pragma unroll
    for (int r = 0; r < 8; r++) {
      int row = rowBase + r;
      float v = acc[t][r];
      if constexpr (EPI == EPI_FCIN) {
        // h = (1-g)*relu(xW+b) + g*c ; write h0 (f32) and Bcat left half (f16)
        float h = fmaxf(v + bias0[col], 0.f);
        h = (1.0f - GAMMA_C) * h + GAMMA_C * bias1[col];
        outF[(size_t)row * NHID + col] = h;
        outH[(size_t)row * (2 * NHID) + col] = (_Float16)h;
      } else if constexpr (EPI == EPI_HW) {
        // hw = h @ w_lin -> Bcat right half
        outH[(size_t)row * (2 * NHID) + NHID + col] = (_Float16)v;
      } else if constexpr (EPI == EPI_BIG) {
        // R = a_hat @ [h | hw]; left: support = (1-a)R + a*h0; right: ahw raw
        if (col < NHID) {
          float s = (1.0f - ALPHA_C) * v + ALPHA_C * aux0[(size_t)row * NHID + col];
          outF[(size_t)row * NHID + col] = s;
          outH[(size_t)row * NHID + col] = (_Float16)s;
        } else {
          outF2[(size_t)row * NHID + (col - NHID)] = v;
        }
      } else if constexpr (EPI == EPI_SMALL) {
        // h_new = (ahw + b_lin) + relu((1-b)*support + b*(support@W) + b_gcnii)
        float sup = aux0[(size_t)row * NHID + col];
        float g = fmaxf((1.0f - p0) * sup + p0 * v + bias1[col], 0.f);
        float hn = aux1[(size_t)row * NHID + col] + bias0[col] + g;
        outH[(size_t)row * (2 * NHID) + col] = (_Float16)hn;
      } else {  // EPI_OUT
        if (col < NCLASS) outF[(size_t)row * NCLASS + col] = v + bias0[col];
      }
    }
  }
}

extern "C" void kernel_launch(void* const* d_in, const int* in_sizes, int n_in,
                              void* d_out, int out_size, void* d_ws, size_t ws_size,
                              hipStream_t stream) {
  const float* x        = (const float*)d_in[0];
  const float* adj      = (const float*)d_in[1];
  const float* fc_in_w  = (const float*)d_in[2];
  const float* fc_in_b  = (const float*)d_in[3];
  const float* cvec     = (const float*)d_in[4];
  const float* w_gcnii  = (const float*)d_in[5];
  const float* b_gcnii  = (const float*)d_in[6];
  const float* w_lin    = (const float*)d_in[7];
  const float* b_lin    = (const float*)d_in[8];
  const float* fc_out_w = (const float*)d_in[9];
  const float* fc_out_b = (const float*)d_in[10];
  float* out = (float*)d_out;

  // workspace carve (256B aligned)
  char* base = (char*)d_ws;
  size_t off = 0;
  auto carve = [&](size_t bytes) { char* p = base + off; off = (off + bytes + 255) & ~(size_t)255; return p; };
  _Float16* ahat   = (_Float16*)carve((size_t)NROW * NROW * 2);     // 128 MB
  float*    dinv   = (float*)   carve((size_t)NROW * 4);
  _Float16* x16    = (_Float16*)carve((size_t)NROW * NFEATP * 2);
  _Float16* w_in16 = (_Float16*)carve((size_t)NFEATP * NHID * 2);
  _Float16* wlin16 = (_Float16*)carve((size_t)NLAYERS * NHID * NHID * 2);
  _Float16* wg16   = (_Float16*)carve((size_t)NLAYERS * NHID * NHID * 2);
  _Float16* fco16  = (_Float16*)carve((size_t)NHID * 64 * 2);
  _Float16* bcat   = (_Float16*)carve((size_t)NROW * 2 * NHID * 2); // [h16 | hw16]
  float*    h0     = (float*)   carve((size_t)NROW * NHID * 4);
  float*    supf   = (float*)   carve((size_t)NROW * NHID * 4);
  _Float16* sup16  = (_Float16*)carve((size_t)NROW * NHID * 2);
  float*    ahw    = (float*)   carve((size_t)NROW * NHID * 4);

  // 1) normalization vector
  k_degree<<<NROW, 256, 0, stream>>>(adj, dinv);
  // 2) f16 normalized adjacency
  k_build_ahat<<<dim3(NROW, 4), 256, 0, stream>>>(adj, dinv, ahat);
  // 3) operand conversions (zero-padded where needed)
  k_cvt<<<(NROW * NFEATP + 255) / 256, 256, 0, stream>>>(x, x16, NROW, NFEAT, NROW, NFEATP);
  k_cvt<<<(NFEATP * NHID + 255) / 256, 256, 0, stream>>>(fc_in_w, w_in16, NFEAT, NHID, NFEATP, NHID);
  k_cvt<<<(NLAYERS * NHID * NHID + 255) / 256, 256, 0, stream>>>(w_lin, wlin16, NLAYERS * NHID, NHID, NLAYERS * NHID, NHID);
  k_cvt<<<(NLAYERS * NHID * NHID + 255) / 256, 256, 0, stream>>>(w_gcnii, wg16, NLAYERS * NHID, NHID, NLAYERS * NHID, NHID);
  k_cvt<<<(NHID * 64 + 255) / 256, 256, 0, stream>>>(fc_out_w, fco16, NHID, NCLASS, NHID, 64);

  // 4) input projection: h0 / Bcat-left = (1-g)relu(x@W+b) + g*c
  k_gemm<EPI_FCIN><<<dim3(NROW / 128, NHID / 64), 256, 0, stream>>>(
      x16, NFEATP, w_in16, NHID, NFEATP,
      h0, nullptr, bcat, fc_in_b, cvec, nullptr, nullptr, 0.f);

  // 5) layers
  for (int i = 0; i < NLAYERS; i++) {
    float beta = logf(0.5f / (float)(i + 1) + 1.0f);
    // hw = h @ w_lin[i] -> Bcat right half
    k_gemm<EPI_HW><<<dim3(NROW / 128, NHID / 64), 256, 0, stream>>>(
        bcat, 2 * NHID, wlin16 + (size_t)i * NHID * NHID, NHID, NHID,
        nullptr, nullptr, bcat, nullptr, nullptr, nullptr, nullptr, 0.f);
    // [ah | ahw] = a_hat @ [h | hw] ; support fused
    k_gemm<EPI_BIG><<<dim3(NROW / 128, (2 * NHID) / 64), 256, 0, stream>>>(
        ahat, NROW, bcat, 2 * NHID, NROW,
        supf, ahw, sup16, nullptr, nullptr, h0, nullptr, 0.f);
    // sw = support @ w_gcnii[i] ; full GCNII epilogue -> Bcat left half (h16)
    k_gemm<EPI_SMALL><<<dim3(NROW / 128, NHID / 64), 256, 0, stream>>>(
        sup16, NHID, wg16 + (size_t)i * NHID * NHID, NHID, NHID,
        nullptr, nullptr, bcat,
        b_lin + (size_t)i * NHID, b_gcnii + (size_t)i * NHID, supf, ahw, beta);
  }

  // 6) output head: logits = h @ fc_out_w + b (N padded to 64, store-guarded)
  k_gemm<EPI_OUT><<<dim3(NROW / 128, 1), 256, 0, stream>>>(
      bcat, 2 * NHID, fco16, 64, NHID,
      out, nullptr, nullptr, fc_out_b, nullptr, nullptr, nullptr, 0.f);

  (void)in_sizes; (void)n_in; (void)out_size; (void)ws_size;
}
